// SuppLayer_89498528514642
// MI455X (gfx1250) — matrix operand order repaired and checked by reading, not compiled, not run on hardware
//
#include <hip/hip_runtime.h>

typedef __attribute__((ext_vector_type(16))) _Float16 v16h;
typedef __attribute__((ext_vector_type(8)))  _Float16 v8h;
typedef __attribute__((ext_vector_type(8)))  float    v8f;
typedef __attribute__((ext_vector_type(4)))  unsigned int u32x4;
typedef __attribute__((ext_vector_type(8)))  int      i32x8;
typedef __attribute__((ext_vector_type(4)))  int      i32x4;
typedef unsigned long long u64;

#define B_DIM 4096
#define K_DIM 4096
#define NCLS  1000
#define NPAD  1024
#define NSUPP 64

#define TM 128
#define TN 256   // 8 waves x 32 columns: 2 B-fragments per wave -> 2 WMMAs per A-fragment load
#define TK 32
#define LDA 40   // LDS A row stride (halves) = 32 data + 8 pad (TDM pad reproduces this)
#define LDB 40

// ---------------------------------------------------------------------------
// TDM: issue a 2D tensor_load_to_lds of a [TM x TK] f16 tile from X_f16
// (row-major, row stride K_DIM halves) into LDS with 16B padding per row.
// Descriptor layout per CDNA5 ISA ch.8 (D# groups 0-3).
// ---------------------------------------------------------------------------
__device__ __forceinline__ void tdm_load_a_tile(const _Float16* gsrc, void* lds_dst) {
  u64 ga = (u64)(void*)gsrc;          // 57-bit global byte address of tile start
  unsigned lds = (unsigned)(u64)lds_dst;

  u32x4 g0;
  g0[0] = 1u;                                              // count=1, user descriptor
  g0[1] = lds;                                             // lds_addr (bytes)
  g0[2] = (unsigned)(ga & 0xFFFFFFFFu);                    // global_addr[31:0]
  g0[3] = (unsigned)((ga >> 32) & 0x01FFFFFFu) | (2u << 30); // global_addr[56:32] | type=2

  i32x8 g1;
  // data_size=1 (2B), pad_enable=1, pad_interval=3 (16 DW = 64B row),
  // pad_amount=3 (4 DW = 16B) -> LDS row stride 80B = LDA halves
  g1[0] = (int)((1u << 16) | (1u << 20) | (3u << 22) | (3u << 25));
  g1[1] = (int)((K_DIM & 0xFFFFu) << 16);                  // tensor_dim0[15:0]<<16
  g1[2] = (int)((K_DIM >> 16) | ((B_DIM & 0xFFFFu) << 16));// dim0[31:16] | dim1[15:0]<<16
  g1[3] = (int)((B_DIM >> 16) | (TK << 16));               // dim1[31:16] | tile_dim0<<16
  g1[4] = (int)(TM);                                       // tile_dim1 | tile_dim2=0
  g1[5] = (int)K_DIM;                                      // tensor_dim0_stride[31:0]
  g1[6] = 0;                                               // stride hi | dim1_stride lo
  g1[7] = 0;

  i32x4 g2 = {0, 0, 0, 0};
  i32x4 g3 = {0, 0, 0, 0};
#if defined(__clang_major__) && __clang_major__ >= 23
  i32x8 g4 = {0, 0, 0, 0, 0, 0, 0, 0};
  __builtin_amdgcn_tensor_load_to_lds(g0, g1, g2, g3, g4, 0);
#else
  __builtin_amdgcn_tensor_load_to_lds(g0, g1, g2, g3, 0);
#endif
}

// ---------------------------------------------------------------------------
// Kernel 1: zero the dense f16 weight matrix W [K_DIM x NPAD].
// ---------------------------------------------------------------------------
__global__ void zero_w_kernel(_Float16* __restrict__ W) {
  int i = blockIdx.x * blockDim.x + threadIdx.x;   // 8 halves per thread
  uint4 z; z.x = 0u; z.y = 0u; z.z = 0u; z.w = 0u;
  ((uint4*)W)[i] = z;
}

// ---------------------------------------------------------------------------
// Kernel 2: scatter wSupp into W: W[chunk_map[c,s], c] += wSupp[c,s].
// One thread per class -> single writer per column -> duplicates accumulate.
// ---------------------------------------------------------------------------
__global__ void scatter_w_kernel(const float* __restrict__ wSupp,
                                 const int* __restrict__ cmap,
                                 _Float16* __restrict__ W) {
  int c = blockIdx.x * blockDim.x + threadIdx.x;
  if (c >= NCLS) return;
  for (int s = 0; s < NSUPP; ++s) {
    int   k = cmap[c * NSUPP + s];
    float w = wSupp[c * NSUPP + s];
    _Float16* p = &W[(size_t)k * NPAD + c];
    *p = (_Float16)((float)(*p) + w);
  }
}

// ---------------------------------------------------------------------------
// Kernel 3: one-shot x f32 -> f16 (so the GEMM A tiles are TDM-copyable).
// ---------------------------------------------------------------------------
__global__ void convert_x_kernel(const float* __restrict__ X,
                                 _Float16* __restrict__ Xh) {
  size_t i = (size_t)(blockIdx.x * blockDim.x + threadIdx.x) * 8;
  float4 a = *(const float4*)(X + i);
  float4 b = *(const float4*)(X + i + 4);
  v8h h;
  h[0] = (_Float16)a.x; h[1] = (_Float16)a.y; h[2] = (_Float16)a.z; h[3] = (_Float16)a.w;
  h[4] = (_Float16)b.x; h[5] = (_Float16)b.y; h[6] = (_Float16)b.z; h[7] = (_Float16)b.w;
  *(v8h*)(Xh + i) = h;
}

// ---------------------------------------------------------------------------
// Kernel 4: feats = x_f16 @ W via v_wmma_f32_16x16x32_f16, exp() epilogue.
// A tiles arrive via double-buffered TDM (tensor_load_to_lds + TENSORcnt).
// Each wave owns 32 N-columns (2 B-fragments): every A-fragment load pair
// feeds 2 independent WMMAs -> half the LDS traffic per matrix op and
// latency hiding across the pair.
// ---------------------------------------------------------------------------
__global__ __launch_bounds__(256) void supp_gemm_wmma(
    const _Float16* __restrict__ Xh,   // [B_DIM, K_DIM] f16 row-major
    const _Float16* __restrict__ W,    // [K_DIM, NPAD]  f16 row-major
    float* __restrict__ out)           // [B_DIM, NCLS]  f32
{
  __shared__ _Float16 ldsA[2][TM * LDA];   // double-buffered TDM destination
  __shared__ _Float16 ldsB[TN * LDB];      // [col][k] transposed

  const int tid  = threadIdx.x;
  const int lane = tid & 31;
  const int wv   = tid >> 5;
  const int bm   = blockIdx.x * TM;
  const int bn   = blockIdx.y * TN;

  v8f acc[8][2];
  #pragma unroll
  for (int i = 0; i < 8; ++i) {
    v8f z = {};
    acc[i][0] = z; acc[i][1] = z;
  }

  const int arow = lane & 15;
  const int akb  = (lane >> 4) * 8;    // A: lanes 0-15 K{0..7,16..23}; 16-31 K{8..15,24..31}
  const int bcol = wv * 32 + (lane & 15);  // this wave's first 16-column group
  const int bkb  = (lane >> 4) * 8;    // B: lanes 0-15 K{0..7,16..23}; 16-31 K{8..15,24..31}

  // Prologue: TDM fetch of the first A tile into buffer 0 (wave 0 only;
  // the scalar branch skips the instruction entirely for waves 1-7).
  if (wv == 0)
    tdm_load_a_tile(Xh + (size_t)bm * K_DIM, &ldsA[0][0]);

  const int NITER = K_DIM / TK;
  for (int it = 0; it < NITER; ++it) {
    const int k0  = it * TK;
    const int cur = it & 1;

    // ---- stage B transposed (all threads): one column per thread ----
    // ldsB[col][k] = W[k0+k][bn+col]; lanes hit adjacent columns -> coalesced.
    {
      int col = tid;                       // 0..255
      const _Float16* gp = W + (size_t)k0 * NPAD + bn + col;
      _Float16* dst = &ldsB[col * LDB];
      #pragma unroll
      for (int i = 0; i < TK; ++i)
        dst[i] = gp[(size_t)i * NPAD];
    }

    // ---- TDM prefetch of next A tile into the other buffer, then wait for
    //      the current one (constant-immediate tensorcnt waits) ----
    if (it + 1 < NITER) {
      if (wv == 0) {
        tdm_load_a_tile(Xh + (size_t)bm * K_DIM + (k0 + TK), &ldsA[cur ^ 1][0]);
        __builtin_amdgcn_s_wait_tensorcnt(1);   // current tile complete
      }
    } else {
      if (wv == 0) __builtin_amdgcn_s_wait_tensorcnt(0);
    }
    __syncthreads();

    // ---- 2 B fragments for this wave's 32 columns ----
    v16h bf[2];
    #pragma unroll
    for (int nf = 0; nf < 2; ++nf) {
      const _Float16* bp = &ldsB[(bcol + nf * 16) * LDB + bkb];
      v8h blo = *(const v8h*)bp;
      v8h bhi = *(const v8h*)(bp + 16);
      #pragma unroll
      for (int i = 0; i < 8; ++i) { bf[nf][i] = blo[i]; bf[nf][i + 8] = bhi[i]; }
    }

    // ---- 8 A fragments, 2 WMMAs each ----
    #pragma unroll
    for (int mf = 0; mf < 8; ++mf) {
      const _Float16* ap = &ldsA[cur][(mf * 16 + arow) * LDA + akb];
      v8h alo = *(const v8h*)ap;
      v8h ahi = *(const v8h*)(ap + 16);
      v16h af;
      #pragma unroll
      for (int i = 0; i < 8; ++i) { af[i] = alo[i]; af[i + 8] = ahi[i]; }
      acc[mf][0] = __builtin_amdgcn_wmma_f32_16x16x32_f16(
          false, af, false, bf[0], (short)0, acc[mf][0], false, false);
      acc[mf][1] = __builtin_amdgcn_wmma_f32_16x16x32_f16(
          false, af, false, bf[1], (short)0, acc[mf][1], false, false);
    }
    __syncthreads();
  }

  // ---- epilogue: exp() and store valid columns ----
  // C/D layout: VGPR r, lanes 0-15 -> M=r, N=lane; lanes 16-31 -> M=r+8.
  const int rowoff = 8 * (lane >> 4);
  #pragma unroll
  for (int nf = 0; nf < 2; ++nf) {
    const int colg = bn + wv * 32 + nf * 16 + (lane & 15);
    if (colg < NCLS) {
      #pragma unroll
      for (int mf = 0; mf < 8; ++mf) {
        #pragma unroll
        for (int r = 0; r < 8; ++r) {
          int row = bm + mf * 16 + rowoff + r;
          out[(size_t)row * NCLS + colg] = __expf(acc[mf][nf][r]);
        }
      }
    }
  }
}

// ---------------------------------------------------------------------------
extern "C" void kernel_launch(void* const* d_in, const int* in_sizes, int n_in,
                              void* d_out, int out_size, void* d_ws, size_t ws_size,
                              hipStream_t stream) {
  (void)in_sizes; (void)n_in; (void)out_size; (void)ws_size;

  const float* x         = (const float*)d_in[0];   // [4096, 4096] f32
  const float* wSupp     = (const float*)d_in[1];   // [1000, 64]   f32
  const int*   chunk_map = (const int*)d_in[2];     // [1000, 64]   i32

  float*    out  = (float*)d_out;                            // [4096, 1000] f32
  _Float16* Wf16 = (_Float16*)d_ws;                          // 8 MB
  _Float16* Xf16 = (_Float16*)((char*)d_ws + (size_t)K_DIM * NPAD * 2); // 32 MB

  // 1) zero W
  zero_w_kernel<<<(K_DIM * NPAD / 8) / 256, 256, 0, stream>>>(Wf16);
  // 2) scatter wSupp into W
  scatter_w_kernel<<<(NCLS + 255) / 256, 256, 0, stream>>>(wSupp, chunk_map, Wf16);
  // 3) convert x to f16 (TDM-friendly A source)
  convert_x_kernel<<<(B_DIM * (K_DIM / 8)) / 256, 256, 0, stream>>>(x, Xf16);
  // 4) WMMA GEMM + exp epilogue
  dim3 grid(B_DIM / TM, NPAD / TN);
  supp_gemm_wmma<<<grid, 256, 0, stream>>>(Xf16, Wf16, out);
}
